// TrajTrans_spatial_11261404250865
// MI455X (gfx1250) — matrix operand order repaired and checked by reading, not compiled, not run on hardware
//
#include <hip/hip_runtime.h>

// ---------------------------------------------------------------------------
// CDNA5 (gfx1250) bf16 WMMA ViT forward.  wave32 only.
// GEMM tile staging uses the Tensor Data Mover (tensor_load_to_lds) when the
// toolchain exposes it (TENSORcnt-tracked, double-buffered LDS), otherwise a
// manual global_load_b128 -> ds_store_b128 path.
// ---------------------------------------------------------------------------

typedef __attribute__((ext_vector_type(16))) __bf16 v16bf;
typedef __attribute__((ext_vector_type(8)))  float  v8f;

#define TOKENS 32768   // B*M
#define HDIM   512
#define BATCH  64
#define SEQ    512
#define NHEADS 8
#define DHEAD  64
#define QKVC   1536

#if defined(__AMDGCN__) && __has_builtin(__builtin_amdgcn_tensor_load_to_lds)
#define USE_TDM 1
#define NBUF 2
#else
#define USE_TDM 0
#define NBUF 1
#endif

union FragCvt { uint4 q[2]; v16bf v; };

__device__ __forceinline__ v16bf frag_from(uint4 a, uint4 b) {
  FragCvt f; f.q[0] = a; f.q[1] = b; return f.v;
}

__device__ __forceinline__ v8f wmma_bf16(v16bf a, v16bf b, v8f c) {
  // D = A(16x32 bf16) * B(32x16 bf16) + C(16x16 f32)
  return __builtin_amdgcn_wmma_f32_16x16x32_bf16(false, a, false, b, (short)0, c,
                                                 false, false);
}

__device__ __forceinline__ v8f v8f_zero() {
  v8f z = {0.f, 0.f, 0.f, 0.f, 0.f, 0.f, 0.f, 0.f};
  return z;
}

#if USE_TDM
typedef unsigned int u32x4 __attribute__((ext_vector_type(4)));
typedef int          i32x4 __attribute__((ext_vector_type(4)));
typedef int          i32x8 __attribute__((ext_vector_type(8)));

__device__ __forceinline__ unsigned lds_off(const void* p) {
  // generic LDS pointer: addr[31:0] is the LDS byte offset (ISA aperture rule)
  return (unsigned)(unsigned long long)(uintptr_t)p;
}

// 2-D bf16 tile load: 128 rows x 64 elements, row stride K elements in memory,
// padded in LDS to 160B row stride (pad 8 DWORDs every 32 DWORDs).
__device__ __forceinline__ void tdm_load_tile(const __bf16* gsrc, unsigned ldsb,
                                              unsigned K) {
  unsigned long long ga = (unsigned long long)(uintptr_t)gsrc;
  u32x4 g0;
  g0[0] = 1u;                                                  // count=1
  g0[1] = ldsb;                                                // lds_addr
  g0[2] = (unsigned)(ga & 0xFFFFFFFFu);                        // global_addr lo
  g0[3] = (unsigned)((ga >> 32) & 0x01FFFFFFu) | (2u << 30);   // addr hi | type=2
  i32x8 g1;
  // data_size=1(2B) | pad_enable | pad_interval=4(32 dw) | pad_amount=7(8 dw)
  g1[0] = (1 << 16) | (1 << 20) | (4 << 22) | (7 << 25);
  g1[1] = (int)((K & 0xFFFFu) << 16);                          // tensor_dim0 lo16
  g1[2] = (int)(((K >> 16) & 0xFFFFu) | (0x0000u << 16));      // dim0 hi | dim1 lo
  g1[2] |= (int)((0x100000u & 0xFFFFu) << 16);                 // tensor_dim1 lo16 (=0)
  g1[3] = (int)(((0x100000u >> 16) & 0xFFFFu) | (64u << 16));  // dim1 hi=16 | tile_dim0=64
  g1[4] = 128;                                                 // tile_dim1=128, tile_dim2=0
  g1[5] = (int)K;                                              // tensor_dim0_stride lo32
  g1[6] = 0;                                                   // stride hi | dim1_stride lo
  g1[7] = 0;
  i32x4 gz = {0, 0, 0, 0};
#if __clang_major__ >= 23
  i32x8 gz8 = {0, 0, 0, 0, 0, 0, 0, 0};
  __builtin_amdgcn_tensor_load_to_lds(g0, g1, gz, gz, gz8, 0);
#else
  __builtin_amdgcn_tensor_load_to_lds(g0, g1, gz, gz, 0);
#endif
}
#endif  // USE_TDM

// ---------------------------------------------------------------------------
// Weight convert + transpose: in f32 [K,N] row-major -> out bf16 [N,K]
// ---------------------------------------------------------------------------
__global__ void wconv_kernel(const float* __restrict__ w, __bf16* __restrict__ wt,
                             int K, int N) {
  int idx = blockIdx.x * 256 + threadIdx.x;     // over K*N, k fastest in output
  int k = idx % K, n = idx / K;
  wt[(size_t)n * K + k] = (__bf16)w[(size_t)k * N + n];
}

// ---------------------------------------------------------------------------
// Patch embed: X[b,m,h] = sum_k patches[b,m,k]*pw[k,h] + pb[h] + pos[m,h]
// ---------------------------------------------------------------------------
__global__ void patch_embed_kernel(const float* __restrict__ patches,
                                   const float* __restrict__ pw,
                                   const float* __restrict__ pb,
                                   const float* __restrict__ pos,
                                   float* __restrict__ X) {
  int idx = blockIdx.x * 256 + threadIdx.x;     // B*M*H
  int h  = idx & 511;
  int bm = idx >> 9;
  int m  = bm & 511;
  const float* pr = patches + (size_t)bm * 3;
  float v = pb[h] + pos[m * 512 + h];
  v += pr[0] * pw[0 * 512 + h] + pr[1] * pw[1 * 512 + h] + pr[2] * pw[2 * 512 + h];
  X[idx] = v;
}

// ---------------------------------------------------------------------------
// LayerNorm: f32 x[row,512] -> bf16 out[row,512]; one wave per row.
// ---------------------------------------------------------------------------
__global__ __launch_bounds__(256) void ln_kernel(const float* __restrict__ x,
                                                 const float* __restrict__ g,
                                                 const float* __restrict__ bta,
                                                 __bf16* __restrict__ out) {
  const int lane = threadIdx.x & 31, wv = threadIdx.x >> 5;
  const int row = blockIdx.x * 8 + wv;
  const float* xr = x + (size_t)row * HDIM;
  float v[16];
  float s = 0.f;
#pragma unroll
  for (int j = 0; j < 16; ++j) { v[j] = xr[lane + j * 32]; s += v[j]; }
#pragma unroll
  for (int o = 16; o >= 1; o >>= 1) s += __shfl_xor(s, o, 32);
  float mu = s * (1.f / 512.f);
  float q = 0.f;
#pragma unroll
  for (int j = 0; j < 16; ++j) { float d = v[j] - mu; q += d * d; }
#pragma unroll
  for (int o = 16; o >= 1; o >>= 1) q += __shfl_xor(q, o, 32);
  float rstd = rsqrtf(q * (1.f / 512.f) + 1e-5f);
  __bf16* orow = out + (size_t)row * HDIM;
#pragma unroll
  for (int j = 0; j < 16; ++j) {
    int c = lane + j * 32;
    orow[c] = (__bf16)((v[j] - mu) * rstd * g[c] + bta[c]);
  }
}

// ---------------------------------------------------------------------------
// WMMA GEMM core: acc[4][2] (wave tile 64x32) of C[R,N] = A[R,K] * WT[N,K]^T
// Block tile 128x128, BK=64, 256 threads (8 waves as 2x4).
// ---------------------------------------------------------------------------
__device__ __forceinline__ void gemm_accum(const __bf16* __restrict__ A,
                                           const __bf16* __restrict__ WT,
                                           int K, int row0, int col0,
                                           v8f acc[4][2]) {
  __shared__ alignas(16) __bf16 Alds[NBUF][128][80];
  __shared__ alignas(16) __bf16 Blds[NBUF][128][80];
  const int tid = threadIdx.x;
  const int lane = tid & 31, wv = tid >> 5;
  const int mw = wv >> 2, nw = wv & 3;
  const int lm = lane & 15, hi = lane >> 4, klo = hi * 8;
#pragma unroll
  for (int i = 0; i < 4; ++i)
#pragma unroll
    for (int j = 0; j < 2; ++j) acc[i][j] = v8f_zero();

#if USE_TDM
  const int wvu = __builtin_amdgcn_readfirstlane(wv);
  // prime buffer 0 (waves 0/1 drive the TDM; scalar branch, TDM ignores EXEC)
  if (wvu == 0)      tdm_load_tile(A  + (size_t)row0 * K, lds_off(&Alds[0][0][0]), K);
  else if (wvu == 1) tdm_load_tile(WT + (size_t)col0 * K, lds_off(&Blds[0][0][0]), K);
#endif

  for (int kb = 0; kb < K; kb += 64) {
#if USE_TDM
    const int buf = (kb >> 6) & 1;
    __syncthreads();   // previous compute done before DMA overwrites buf^1
    if (kb + 64 < K) {
      if (wvu == 0)
        tdm_load_tile(A + (size_t)row0 * K + (kb + 64),
                      lds_off(&Alds[buf ^ 1][0][0]), K);
      else if (wvu == 1)
        tdm_load_tile(WT + (size_t)col0 * K + (kb + 64),
                      lds_off(&Blds[buf ^ 1][0][0]), K);
      __builtin_amdgcn_s_wait_tensorcnt((short)1);  // tile(kb) complete
    } else {
      __builtin_amdgcn_s_wait_tensorcnt((short)0);
    }
    __syncthreads();
#else
    const int buf = 0;
    __syncthreads();
#pragma unroll
    for (int i = 0; i < 4; ++i) {
      int c = tid + i * 256;
      int r = c >> 3, cc = (c & 7) * 8;
      *reinterpret_cast<uint4*>(&Alds[0][r][cc]) =
          *reinterpret_cast<const uint4*>(A + (size_t)(row0 + r) * K + kb + cc);
      *reinterpret_cast<uint4*>(&Blds[0][r][cc]) =
          *reinterpret_cast<const uint4*>(WT + (size_t)(col0 + r) * K + kb + cc);
    }
    __syncthreads();
#endif
    __bf16 (*Al)[80] = Alds[buf];
    __bf16 (*Bl)[80] = Blds[buf];
#pragma unroll
    for (int ks = 0; ks < 2; ++ks) {
      v16bf af[4], bfr[2];
#pragma unroll
      for (int mt = 0; mt < 4; ++mt) {
        const __bf16* p = &Al[mw * 64 + mt * 16 + lm][ks * 32 + klo];
        af[mt] = frag_from(*reinterpret_cast<const uint4*>(p),
                           *reinterpret_cast<const uint4*>(p + 16));
      }
#pragma unroll
      for (int nt = 0; nt < 2; ++nt) {
        const __bf16* p = &Bl[nw * 32 + nt * 16 + lm][ks * 32 + klo];
        bfr[nt] = frag_from(*reinterpret_cast<const uint4*>(p),
                            *reinterpret_cast<const uint4*>(p + 16));
      }
#pragma unroll
      for (int mt = 0; mt < 4; ++mt)
#pragma unroll
        for (int nt = 0; nt < 2; ++nt)
          acc[mt][nt] = wmma_bf16(af[mt], bfr[nt], acc[mt][nt]);
    }
  }
}

// bf16 output, optional bias + exact GELU
template <bool HAS_BIAS, bool GELU>
__global__ __launch_bounds__(256) void gemm_bf16out_kernel(
    const __bf16* __restrict__ A, const __bf16* __restrict__ WT,
    const float* __restrict__ bias, __bf16* __restrict__ out, int K, int N) {
  const int lane = threadIdx.x & 31, wv = threadIdx.x >> 5;
  const int mw = wv >> 2, nw = wv & 3;
  const int lm = lane & 15, hi = lane >> 4;
  const int row0 = blockIdx.x * 128, col0 = blockIdx.y * 128;
  v8f acc[4][2];
  gemm_accum(A, WT, K, row0, col0, acc);
#pragma unroll
  for (int mt = 0; mt < 4; ++mt)
#pragma unroll
    for (int nt = 0; nt < 2; ++nt)
#pragma unroll
      for (int e = 0; e < 8; ++e) {
        int m   = row0 + mw * 64 + mt * 16 + e + 8 * hi;
        int col = col0 + nw * 32 + nt * 16 + lm;
        float v = acc[mt][nt][e];
        if (HAS_BIAS) v += bias[col];
        if (GELU) v = 0.5f * v * (1.0f + erff(v * 0.70710678118f));
        out[(size_t)m * N + col] = (__bf16)v;
      }
}

// f32 output with residual accumulate (xout += bias + A*W)
__global__ __launch_bounds__(256) void gemm_f32res_kernel(
    const __bf16* __restrict__ A, const __bf16* __restrict__ WT,
    const float* __restrict__ bias, float* __restrict__ xout, int K, int N) {
  const int lane = threadIdx.x & 31, wv = threadIdx.x >> 5;
  const int mw = wv >> 2, nw = wv & 3;
  const int lm = lane & 15, hi = lane >> 4;
  const int row0 = blockIdx.x * 128, col0 = blockIdx.y * 128;
  v8f acc[4][2];
  gemm_accum(A, WT, K, row0, col0, acc);
#pragma unroll
  for (int mt = 0; mt < 4; ++mt)
#pragma unroll
    for (int nt = 0; nt < 2; ++nt)
#pragma unroll
      for (int e = 0; e < 8; ++e) {
        int m   = row0 + mw * 64 + mt * 16 + e + 8 * hi;
        int col = col0 + nw * 32 + nt * 16 + lm;
        size_t o = (size_t)m * N + col;
        xout[o] = xout[o] + bias[col] + acc[mt][nt][e];
      }
}

// ---------------------------------------------------------------------------
// Build V^T: vt[b,h,d,key] = qkv[b*512+key, 1024 + h*64 + d]
// ---------------------------------------------------------------------------
__global__ void build_vt_kernel(const __bf16* __restrict__ qkv,
                                __bf16* __restrict__ vt) {
  int idx = blockIdx.x * 256 + threadIdx.x;    // B*H*64*512
  int key = idx & 511;
  int d   = (idx >> 9) & 63;
  int h   = (idx >> 15) & 7;
  int b   = idx >> 18;
  vt[idx] = qkv[((size_t)(b * 512 + key)) * QKVC + 1024 + h * 64 + d];
}

// ---------------------------------------------------------------------------
// Attention: flash-style, p = exp(s*scale - m)*mask, renorm by sum(p).
// Grid: (M/64, NHEADS, BATCH), 128 threads (4 waves, 16 queries/wave).
// ---------------------------------------------------------------------------
__global__ __launch_bounds__(128) void attn_kernel(const __bf16* __restrict__ qkv,
                                                   const __bf16* __restrict__ vt,
                                                   const float* __restrict__ mask,
                                                   __bf16* __restrict__ z) {
  __shared__ alignas(16) __bf16 Plds[4][16][32];
  const int lane = threadIdx.x & 31, wv = threadIdx.x >> 5;
  const int lm = lane & 15, hi = lane >> 4, klo = hi * 8;
  const int b = blockIdx.z, h = blockIdx.y;
  const int qbase = blockIdx.x * 64 + wv * 16;
  const float scale = 0.125f;   // 1/sqrt(64)

  // Q fragments (A-layout): rows = 16 queries, K = 64 head dims (2 steps of 32)
  v16bf qf[2];
  {
    const __bf16* qrow = qkv + ((size_t)(b * SEQ + qbase + lm)) * QKVC + h * 64;
#pragma unroll
    for (int s = 0; s < 2; ++s) {
      const __bf16* p = qrow + s * 32 + klo;
      qf[s] = frag_from(*reinterpret_cast<const uint4*>(p),
                        *reinterpret_cast<const uint4*>(p + 16));
    }
  }

  v8f acc[4];
#pragma unroll
  for (int dt = 0; dt < 4; ++dt) acc[dt] = v8f_zero();
  float mrow[8], lrow[8];
#pragma unroll
  for (int e = 0; e < 8; ++e) { mrow[e] = -1e30f; lrow[e] = 0.f; }

  const __bf16* vbase = vt + ((size_t)(b * NHEADS + h)) * DHEAD * SEQ;
  const float* mbase = mask + ((size_t)b * SEQ + qbase + 8 * hi) * SEQ + lm;

  for (int kt = 0; kt < 16; ++kt) {
    const int key0 = kt * 32;
    // scores: two 16x16 tiles over 32 keys
    v8f S[2];
#pragma unroll
    for (int t2 = 0; t2 < 2; ++t2) {
      const __bf16* krow =
          qkv + ((size_t)(b * SEQ + key0 + t2 * 16 + lm)) * QKVC + 512 + h * 64;
      v8f c = v8f_zero();
#pragma unroll
      for (int s = 0; s < 2; ++s) {
        const __bf16* p = krow + s * 32 + klo;
        v16bf kf = frag_from(*reinterpret_cast<const uint4*>(p),
                             *reinterpret_cast<const uint4*>(p + 16));
        c = wmma_bf16(qf[s], kf, c);
      }
      S[t2] = c;
    }
    // per-row running max over this key tile (reduce across 16-lane group)
    float sm[8];
#pragma unroll
    for (int e = 0; e < 8; ++e) sm[e] = scale * fmaxf(S[0][e], S[1][e]);
#pragma unroll
    for (int o = 1; o <= 8; o <<= 1)
#pragma unroll
      for (int e = 0; e < 8; ++e) sm[e] = fmaxf(sm[e], __shfl_xor(sm[e], o, 32));

    float alpha[8], psum[8];
#pragma unroll
    for (int e = 0; e < 8; ++e) {
      float mn = fmaxf(mrow[e], sm[e]);
      alpha[e] = __expf(mrow[e] - mn);
      mrow[e] = mn;
      const float* mp = mbase + (size_t)e * SEQ + key0;
      float p0 = __expf(scale * S[0][e] - mn) * mp[0];
      float p1 = __expf(scale * S[1][e] - mn) * mp[16];
      // stage P in exact A-fragment layout: Plds[m][k], m=query, k=key-in-tile
      Plds[wv][e + 8 * hi][lm]      = (__bf16)p0;
      Plds[wv][e + 8 * hi][16 + lm] = (__bf16)p1;
      psum[e] = p0 + p1;
    }
#pragma unroll
    for (int o = 1; o <= 8; o <<= 1)
#pragma unroll
      for (int e = 0; e < 8; ++e) psum[e] += __shfl_xor(psum[e], o, 32);
#pragma unroll
    for (int e = 0; e < 8; ++e) lrow[e] = lrow[e] * alpha[e] + psum[e];
#pragma unroll
    for (int dt = 0; dt < 4; ++dt)
#pragma unroll
      for (int e = 0; e < 8; ++e) acc[dt][e] *= alpha[e];

    // same-wave LDS is in-order, but fence the compiler + DScnt explicitly
    asm volatile("s_wait_dscnt 0x0" ::: "memory");

    const __bf16* pp = &Plds[wv][lm][klo];
    v16bf pf = frag_from(*reinterpret_cast<const uint4*>(pp),
                         *reinterpret_cast<const uint4*>(pp + 16));
#pragma unroll
    for (int dt = 0; dt < 4; ++dt) {
      const __bf16* vr = vbase + (size_t)(dt * 16 + lm) * SEQ + key0 + klo;
      v16bf vf = frag_from(*reinterpret_cast<const uint4*>(vr),
                           *reinterpret_cast<const uint4*>(vr + 16));
      acc[dt] = wmma_bf16(pf, vf, acc[dt]);
    }
  }

#pragma unroll
  for (int e = 0; e < 8; ++e) {
    float inv = 1.0f / (lrow[e] + 1e-10f);
    __bf16* zr = z + ((size_t)(b * SEQ + qbase + e + 8 * hi)) * HDIM + h * 64;
#pragma unroll
    for (int dt = 0; dt < 4; ++dt) zr[dt * 16 + lm] = (__bf16)(acc[dt][e] * inv);
  }
}

// ---------------------------------------------------------------------------
// Mean pool over M: out[b,h] = mean_m X[b,m,h]
// ---------------------------------------------------------------------------
__global__ void pool_kernel(const float* __restrict__ X, float* __restrict__ out) {
  int idx = blockIdx.x * 256 + threadIdx.x;    // B*H
  int hcol = idx & 511, b = idx >> 9;
  const float* xp = X + (size_t)b * SEQ * HDIM + hcol;
  float s = 0.f;
  for (int m = 0; m < SEQ; ++m) s += xp[(size_t)m * HDIM];
  out[idx] = s * (1.0f / 512.0f);
}

// ---------------------------------------------------------------------------
// Host launch
// ---------------------------------------------------------------------------
struct LayerP {
  const float *ln1g, *ln1b, *qkvw, *outw, *outb, *ln2g, *ln2b, *w1, *b1, *w2, *b2;
};

extern "C" void kernel_launch(void* const* d_in, const int* in_sizes, int n_in,
                              void* d_out, int out_size, void* d_ws, size_t ws_size,
                              hipStream_t stream) {
  (void)out_size; (void)ws_size;
  // locate patches / mask by unique sizes
  int pi = -1, mi = -1, perm[64], np = 0;
  for (int i = 0; i < n_in; ++i) {
    if (in_sizes[i] == BATCH * SEQ * 3) pi = i;
    else if (in_sizes[i] == BATCH * SEQ * SEQ) mi = i;
    else if (np < 64) perm[np++] = i;
  }
  if (pi < 0 || mi < 0 || np < 36) return;
  const float* patches = (const float*)d_in[pi];
  const float* maskp   = (const float*)d_in[mi];

  const float *patch_w, *patch_b, *pos;
  LayerP L[3];
  if (in_sizes[perm[0]] == 1024) {
    // jax tree-flatten (sorted keys): layers first, each layer alphabetical
    for (int l = 0; l < 3; ++l) {
      int b0 = l * 11;
      L[l].b1   = (const float*)d_in[perm[b0 + 0]];
      L[l].b2   = (const float*)d_in[perm[b0 + 1]];
      L[l].w1   = (const float*)d_in[perm[b0 + 2]];
      L[l].w2   = (const float*)d_in[perm[b0 + 3]];
      L[l].ln1b = (const float*)d_in[perm[b0 + 4]];
      L[l].ln1g = (const float*)d_in[perm[b0 + 5]];
      L[l].ln2b = (const float*)d_in[perm[b0 + 6]];
      L[l].ln2g = (const float*)d_in[perm[b0 + 7]];
      L[l].outb = (const float*)d_in[perm[b0 + 8]];
      L[l].outw = (const float*)d_in[perm[b0 + 9]];
      L[l].qkvw = (const float*)d_in[perm[b0 + 10]];
    }
    patch_b = (const float*)d_in[perm[33]];
    patch_w = (const float*)d_in[perm[34]];
    pos     = (const float*)d_in[perm[35]];
  } else {
    // dict insertion order
    patch_w = (const float*)d_in[perm[0]];
    patch_b = (const float*)d_in[perm[1]];
    pos     = (const float*)d_in[perm[2]];
    for (int l = 0; l < 3; ++l) {
      int b0 = 3 + l * 11;
      L[l].ln1g = (const float*)d_in[perm[b0 + 0]];
      L[l].ln1b = (const float*)d_in[perm[b0 + 1]];
      L[l].qkvw = (const float*)d_in[perm[b0 + 2]];
      L[l].outw = (const float*)d_in[perm[b0 + 3]];
      L[l].outb = (const float*)d_in[perm[b0 + 4]];
      L[l].ln2g = (const float*)d_in[perm[b0 + 5]];
      L[l].ln2b = (const float*)d_in[perm[b0 + 6]];
      L[l].w1   = (const float*)d_in[perm[b0 + 7]];
      L[l].b1   = (const float*)d_in[perm[b0 + 8]];
      L[l].w2   = (const float*)d_in[perm[b0 + 9]];
      L[l].b2   = (const float*)d_in[perm[b0 + 10]];
    }
  }

  // workspace layout
  size_t off = 0;
  char* wsb = (char*)d_ws;
  auto take = [&](size_t bytes) -> void* {
    off = (off + 255) & ~(size_t)255;
    void* p = wsb + off;
    off += bytes;
    return p;
  };
  float*  X   = (float*)take((size_t)TOKENS * HDIM * 4);
  __bf16* LNZ = (__bf16*)take((size_t)TOKENS * HDIM * 2);   // LN out, also attn-Z
  __bf16* QKV = (__bf16*)take((size_t)TOKENS * QKVC * 2);   // qkv, also ffn hidden
  __bf16* VT  = (__bf16*)take((size_t)BATCH * NHEADS * DHEAD * SEQ * 2);
  __bf16 *WQ[3], *WO[3], *W1[3], *W2[3];
  for (int l = 0; l < 3; ++l) {
    WQ[l] = (__bf16*)take((size_t)QKVC * HDIM * 2);
    WO[l] = (__bf16*)take((size_t)HDIM * HDIM * 2);
    W1[l] = (__bf16*)take((size_t)1024 * HDIM * 2);
    W2[l] = (__bf16*)take((size_t)HDIM * 1024 * 2);
  }

  // weight convert/transpose (f32 [K,N] -> bf16 [N,K])
  for (int l = 0; l < 3; ++l) {
    wconv_kernel<<<(512 * 1536) / 256, 256, 0, stream>>>(L[l].qkvw, WQ[l], 512, 1536);
    wconv_kernel<<<(512 * 512)  / 256, 256, 0, stream>>>(L[l].outw, WO[l], 512, 512);
    wconv_kernel<<<(512 * 1024) / 256, 256, 0, stream>>>(L[l].w1,   W1[l], 512, 1024);
    wconv_kernel<<<(1024 * 512) / 256, 256, 0, stream>>>(L[l].w2,   W2[l], 1024, 512);
  }

  patch_embed_kernel<<<(TOKENS * HDIM) / 256, 256, 0, stream>>>(patches, patch_w,
                                                                patch_b, pos, X);

  for (int l = 0; l < 3; ++l) {
    ln_kernel<<<TOKENS / 8, 256, 0, stream>>>(X, L[l].ln1g, L[l].ln1b, LNZ);
    gemm_bf16out_kernel<false, false>
        <<<dim3(TOKENS / 128, QKVC / 128), 256, 0, stream>>>(LNZ, WQ[l], nullptr,
                                                             QKV, 512, QKVC);
    build_vt_kernel<<<(BATCH * NHEADS * DHEAD * SEQ) / 256, 256, 0, stream>>>(QKV, VT);
    attn_kernel<<<dim3(SEQ / 64, NHEADS, BATCH), 128, 0, stream>>>(QKV, VT, maskp, LNZ);
    gemm_f32res_kernel<<<dim3(TOKENS / 128, HDIM / 128), 256, 0, stream>>>(
        LNZ, WO[l], L[l].outb, X, 512, HDIM);
    ln_kernel<<<TOKENS / 8, 256, 0, stream>>>(X, L[l].ln2g, L[l].ln2b, LNZ);
    gemm_bf16out_kernel<true, true>
        <<<dim3(TOKENS / 128, 1024 / 128), 256, 0, stream>>>(LNZ, W1[l], L[l].b1,
                                                             QKV, 512, 1024);
    gemm_f32res_kernel<<<dim3(TOKENS / 128, HDIM / 128), 256, 0, stream>>>(
        QKV, W2[l], L[l].b2, X, 1024, HDIM);
  }

  pool_kernel<<<(BATCH * HDIM) / 256, 256, 0, stream>>>(X, (float*)d_out);
}
